// TbpNet_44332652429561
// MI455X (gfx1250) — matrix operand order repaired and checked by reading, not compiled, not run on hardware
//
#include <hip/hip_runtime.h>
#include <hip/hip_bf16.h>
#include <math.h>

// Problem constants (match reference)
#define F     128
#define KB    25
#define NL    3
#define BB    16
#define AA    256
#define NBH   64
#define MROWS (BB*AA)        // 4096 atom rows
#define EDGES (BB*AA*NBH)    // 262144 edges

typedef __bf16 bf16;
typedef __attribute__((ext_vector_type(16))) __bf16 v16bf;
typedef __attribute__((ext_vector_type(8)))  float  v8f;

union BF16Frag { v16bf v; uint4 q[2]; };

__device__ __forceinline__ float sspf(float x) {
  // shifted softplus: log1p(exp(x)) - ln(2)
  float sp = (x > 20.0f) ? x : log1pf(__expf(x));
  return sp - 0.69314718055994531f;
}

// ---------------------------------------------------------------------------
// One 16x16 WMMA output tile, D = A(16xKTOT) * B(KTOTx16) + acc, bf16->f32.
// A: row-major [16 x lda], BT: B transposed, row-major [16 x ldbt] (cols of B).
// Fragment layouts per CDNA5 ISA 7.12.2:
//   A 16x32 bf16: lane m=lane&15, half=lane>>4; elems 0..7 = K[8h..8h+7],
//                 elems 8..15 = K[16+8h..23+8h]  -> two contiguous 16B chunks.
//   B 32x16 bf16: lane n=lane&15, half h; same K order as A -> two 16B chunks.
// ---------------------------------------------------------------------------
template<int KTOT>
__device__ __forceinline__ v8f wmma_tile_bf16(const bf16* __restrict__ A, int lda,
                                              const bf16* __restrict__ BT, int ldbt,
                                              v8f acc) {
  const int lane = threadIdx.x & 31;
  const int rc   = lane & 15;          // row of A tile / col of B tile
  const int hf   = (lane >> 4) & 1;
#pragma unroll
  for (int k0 = 0; k0 < KTOT; k0 += 32) {
    BF16Frag af, bfr;
    const bf16* pa = A + (size_t)rc * lda + k0;
    af.q[0] = *(const uint4*)(pa + 8 * hf);
    af.q[1] = *(const uint4*)(pa + 16 + 8 * hf);
    const bf16* pb = BT + (size_t)rc * ldbt + k0 + 8 * hf;
    bfr.q[0] = *(const uint4*)(pb);
    bfr.q[1] = *(const uint4*)(pb + 16);
    if (k0 + 32 < KTOT) {
      __builtin_prefetch((const void*)(pa + 32), 0, 1);   // global_prefetch_b8
      __builtin_prefetch((const void*)(pb + 32), 0, 1);
    }
    acc = __builtin_amdgcn_wmma_f32_16x16x32_bf16(false, af.v, false, bfr.v,
                                                  (short)0, acc, false, false);
  }
  return acc;
}

// ---------------------------------------------------------------------------
// Weight prep: f32 [L,K,N] -> bf16 transposed [L,N,K] (B^T layout for WMMA)
// ---------------------------------------------------------------------------
__global__ void prep_sq_kernel(const float* __restrict__ W, bf16* __restrict__ WT) {
  size_t i = (size_t)blockIdx.x * 256 + threadIdx.x;      // L*F*F threads
  int l = (int)(i / (F * F)); int rem = (int)(i % (F * F));
  int k = rem / F, n = rem % F;
  WT[(size_t)l * F * F + (size_t)n * F + k] = (bf16)W[i];
}

__global__ void prep_fs_kernel(const float* __restrict__ W, bf16* __restrict__ WT) {
  size_t i = (size_t)blockIdx.x * 256 + threadIdx.x;      // L*F*32 threads
  int l = (int)(i / (F * 32)); int rem = (int)(i % (F * 32));
  int n = rem / 32, k = rem % 32;
  float v = (k < KB) ? W[(size_t)l * KB * F + (size_t)k * F + n] : 0.0f;
  WT[i] = (bf16)v;                                        // [L][n][k], K padded to 32
}

// ---------------------------------------------------------------------------
// Embedding lookup: x = emb[Z]
// ---------------------------------------------------------------------------
__global__ void embed_kernel(const float* __restrict__ emb, const int* __restrict__ Z,
                             float* __restrict__ x, bf16* __restrict__ xbf) {
  const int atom = blockIdx.x; const int c = threadIdx.x;
  float v = emb[(size_t)Z[atom] * F + c];
  x[(size_t)atom * F + c]   = v;
  xbf[(size_t)atom * F + c] = (bf16)v;
}

// ---------------------------------------------------------------------------
// Per-edge geometry + Bernstein basis (layer-independent, computed once).
// Writes: dhat [E,4] f32, fbf [E,32] bf16 = bernstein(r)*fc (fc folded in,
// padded K 25->32 with zeros so the filter GEMM is exactly one WMMA deep).
// ---------------------------------------------------------------------------
__global__ void __launch_bounds__(256)
geom_kernel(const float* __restrict__ pos, const float* __restrict__ nmask,
            const int* __restrict__ nbrs, float* __restrict__ dhat,
            bf16* __restrict__ fbf) {
  const size_t e = (size_t)blockIdx.x * 256 + threadIdx.x;
  const size_t atom = e >> 6;                 // b*AA + a
  const int bb = (int)(atom >> 8);
  const int j = nbrs[e];
  float px = pos[atom * 3 + 0], py = pos[atom * 3 + 1], pz = pos[atom * 3 + 2];
  size_t ja = (size_t)bb * AA + (size_t)j;
  float dx = pos[ja * 3 + 0] - px, dy = pos[ja * 3 + 1] - py, dz = pos[ja * 3 + 2] - pz;
  float msk = nmask[e];
  float r = sqrtf(dx * dx + dy * dy + dz * dz + 1e-12f) * msk;
  float inv = 1.0f / (r + 1e-9f);
  dhat[e * 4 + 0] = dx * inv; dhat[e * 4 + 1] = dy * inv; dhat[e * 4 + 2] = dz * inv;
  float fc = (r < 5.0f) ? 0.5f * (__cosf(r * (3.14159265358979f / 5.0f)) + 1.0f) : 0.0f;
  fc *= msk;
  float xx  = __expf(-r);
  float lx  = __logf(xx + 1e-10f);
  float l1x = __logf(1.0f - xx + 1e-10f);
  const float lgn = lgammaf((float)KB);       // gammaln(n+1), n = 24
#pragma unroll 4
  for (int k = 0; k < 32; ++k) {
    float v = 0.0f;
    if (k < KB) {
      float lb = lgn - lgammaf((float)k + 1.0f) - lgammaf((float)(KB - 1 - k) + 1.0f);
      v = __expf(lb + (float)k * lx + (float)(KB - 1 - k) * l1x) * fc;
    }
    fbf[e * 32 + k] = (bf16)v;
  }
}

// ---------------------------------------------------------------------------
// Generic feature GEMM: C[4096x128] = act(rowscale * (A_bf16 @ W^T_bf16) + bias)
// One block = one 16-row M tile; 8 waves cover the 8 N tiles. K=128 -> 4 WMMA.
// Optional outputs: f32, bf16, and batch-transposed bf16 ([B][F][A], for vv).
// ---------------------------------------------------------------------------
__global__ void __launch_bounds__(256)
gemm_feat_kernel(const bf16* __restrict__ Abf, const bf16* __restrict__ WT,
                 const float* __restrict__ bias, const float* __restrict__ rowscale,
                 int act, float* __restrict__ Cf, bf16* __restrict__ Cbf,
                 bf16* __restrict__ CbfT) {
  const int mt = blockIdx.x;
  const int nt = threadIdx.x >> 5;
  const int lane = threadIdx.x & 31;
  const int hf = (lane >> 4) & 1;
  const int col = nt * 16 + (lane & 15);
  v8f acc = {};
  acc = wmma_tile_bf16<F>(Abf + (size_t)mt * 16 * F, F,
                          WT + (size_t)nt * 16 * F, F, acc);
  const float b = bias ? bias[col] : 0.0f;
#pragma unroll
  for (int r = 0; r < 8; ++r) {
    int m = mt * 16 + r + 8 * hf;               // D tile: row r+8*half, col lane&15
    float val = acc[r];
    if (rowscale) val *= rowscale[m];           // == (mask*x)@W, linear fold
    val += b;
    if (act) val = sspf(val);
    if (Cf)  Cf[(size_t)m * F + col] = val;
    if (Cbf) Cbf[(size_t)m * F + col] = (bf16)val;
    if (CbfT) {
      int batch = m >> 8, al = m & (AA - 1);
      CbfT[((size_t)batch * F + col) * AA + al] = (bf16)val;
    }
  }
}

// ---------------------------------------------------------------------------
// Fused local interaction: per atom (block), per neighbor-tile WMMA filter
// generation consumed in-register by the gather/scatter-sum. Never spills
// Ws/Wp (2 x 134 MB/layer) to memory.
// Block = 256 threads = 8 waves; wave nt owns output cols [16nt,16nt+16).
// ---------------------------------------------------------------------------
__global__ void __launch_bounds__(256)
edge_local_kernel(const bf16* __restrict__ fbf, const float* __restrict__ dhat,
                  const int* __restrict__ nbrs, const float* __restrict__ y,
                  const bf16* __restrict__ WfsT, const bf16* __restrict__ WfpT,
                  bf16* __restrict__ cbf) {
  const int atom = blockIdx.x;                  // b*AA + a
  const int bb = atom >> 8;
  __shared__ int   snb[NBH];
  __shared__ float sdh[NBH * 4];
  const int t = threadIdx.x;
  if (t < NBH) snb[t] = nbrs[(size_t)atom * NBH + t];
  sdh[t] = dhat[(size_t)atom * NBH * 4 + t];    // 256 floats
  __syncthreads();
  const int nt = t >> 5, lane = t & 31, hf = (lane >> 4) & 1;
  const int col = nt * 16 + (lane & 15);
  const bf16* fA = fbf + (size_t)atom * NBH * 32;
  const bf16* bs = WfsT + (size_t)nt * 16 * 32;
  const bf16* bp = WfpT + (size_t)nt * 16 * 32;
  float cs = 0.f, t0 = 0.f, t1 = 0.f, t2 = 0.f;
#pragma unroll
  for (int mt = 0; mt < 4; ++mt) {              // 4 tiles of 16 neighbors
    v8f z0 = {}; v8f z1 = {};
    v8f Ws = wmma_tile_bf16<32>(fA + mt * 16 * 32, 32, bs, 32, z0);
    v8f Wp = wmma_tile_bf16<32>(fA + mt * 16 * 32, 32, bp, 32, z1);
#pragma unroll
    for (int r = 0; r < 8; ++r) {
      int n = mt * 16 + r + 8 * hf;             // neighbor slot for this D elem
      float yj = y[((size_t)bb * AA + snb[n]) * F + col];   // L2-resident gather
      cs += yj * Ws[r];
      float w = yj * Wp[r];
      t0 += sdh[n * 4 + 0] * w;
      t1 += sdh[n * 4 + 1] * w;
      t2 += sdh[n * 4 + 2] * w;
    }
  }
  // combine the two half-wave partials (wave32: xor lane 16)
  cs += __shfl_xor(cs, 16, 32);
  t0 += __shfl_xor(t0, 16, 32);
  t1 += __shfl_xor(t1, 16, 32);
  t2 += __shfl_xor(t2, 16, 32);
  if (hf == 0)
    cbf[(size_t)atom * F + col] = (bf16)(cs + t0 * t0 + t1 * t1 + t2 * t2);
}

// ---------------------------------------------------------------------------
// Attention logits: S = q @ k^T / sqrt(F) + colmask. k_bf [A,F] is already B^T.
// ---------------------------------------------------------------------------
__global__ void __launch_bounds__(256)
att_logits_kernel(const bf16* __restrict__ qbf, const bf16* __restrict__ kbf,
                  const float* __restrict__ amask, float* __restrict__ logits) {
  const int bb = blockIdx.y;
  const int tile = blockIdx.x * 8 + (threadIdx.x >> 5);    // 256 tiles / batch
  const int mt = tile >> 4, nt = tile & 15;
  const int lane = threadIdx.x & 31, hf = (lane >> 4) & 1;
  const int c = nt * 16 + (lane & 15);
  v8f acc = {};
  acc = wmma_tile_bf16<F>(qbf + ((size_t)bb * AA + mt * 16) * F, F,
                          kbf + ((size_t)bb * AA + nt * 16) * F, F, acc);
  const float sc = 0.0883883476483184f;                    // 1/sqrt(128)
  const float madd = (1.0f - amask[(size_t)bb * AA + c]) * (-1e9f);
#pragma unroll
  for (int r = 0; r < 8; ++r) {
    int m = mt * 16 + r + 8 * hf;
    logits[((size_t)bb * AA + m) * AA + c] = acc[r] * sc + madd;
  }
}

// ---------------------------------------------------------------------------
// Row softmax (256 cols), output bf16 probs for the P@vv WMMA.
// ---------------------------------------------------------------------------
__global__ void __launch_bounds__(256)
softmax_kernel(const float* __restrict__ logits, bf16* __restrict__ P) {
  const size_t row = blockIdx.x;
  const int t = threadIdx.x;
  __shared__ float red[AA];
  float v = logits[row * AA + t];
  red[t] = v; __syncthreads();
  for (int s = 128; s > 0; s >>= 1) { if (t < s) red[t] = fmaxf(red[t], red[t + s]); __syncthreads(); }
  float mx = red[0]; __syncthreads();
  float e = __expf(v - mx);
  red[t] = e; __syncthreads();
  for (int s = 128; s > 0; s >>= 1) { if (t < s) red[t] += red[t + s]; __syncthreads(); }
  P[row * AA + t] = (bf16)(e / red[0]);
}

// ---------------------------------------------------------------------------
// n = (P @ vv) * mask, fused residual: x_out = x + v + n (+ bf16 mirror).
// vvT [B][F][A] gives contiguous B^T fragments. K=256 -> 8 WMMA.
// ---------------------------------------------------------------------------
__global__ void __launch_bounds__(256)
att_out_kernel(const bf16* __restrict__ P, const bf16* __restrict__ vvT,
               const float* __restrict__ amask, const float* __restrict__ xin,
               const float* __restrict__ vres, float* __restrict__ xout,
               bf16* __restrict__ xbf) {
  const int bb = blockIdx.y;
  const int mt = blockIdx.x;
  const int nt = threadIdx.x >> 5;
  const int lane = threadIdx.x & 31, hf = (lane >> 4) & 1;
  const int col = nt * 16 + (lane & 15);
  v8f acc = {};
  acc = wmma_tile_bf16<AA>(P + ((size_t)bb * AA + mt * 16) * AA, AA,
                           vvT + ((size_t)bb * F + nt * 16) * AA, AA, acc);
#pragma unroll
  for (int r = 0; r < 8; ++r) {
    int m = mt * 16 + r + 8 * hf;
    size_t g = (size_t)bb * AA + m;
    float nv = acc[r] * amask[g];
    float xv = xin[g * F + col] + vres[g * F + col] + nv;
    xout[g * F + col] = xv;
    xbf[g * F + col]  = (bf16)xv;
  }
}

// ---------------------------------------------------------------------------
extern "C" void kernel_launch(void* const* d_in, const int* in_sizes, int n_in,
                              void* d_out, int out_size, void* d_ws, size_t ws_size,
                              hipStream_t stream) {
  const float* positions = (const float*)d_in[0];
  const float* nmask     = (const float*)d_in[1];
  const float* amask     = (const float*)d_in[2];
  const float* emb       = (const float*)d_in[3];
  const float* Wfs       = (const float*)d_in[4];
  const float* Wfp       = (const float*)d_in[5];
  const float* Win       = (const float*)d_in[6];
  const float* Wf2o      = (const float*)d_in[7];
  const float* bf2o      = (const float*)d_in[8];
  const float* Wd        = (const float*)d_in[9];
  const float* bd        = (const float*)d_in[10];
  const float* Wq        = (const float*)d_in[11];
  const float* Wk        = (const float*)d_in[12];
  const float* Wv        = (const float*)d_in[13];
  const int*   Z         = (const int*)d_in[14];
  const int*   nbrs      = (const int*)d_in[15];
  float* out = (float*)d_out;

  char* base = (char*)d_ws;
  size_t off = 0;
  auto take = [&](size_t bytes) -> void* {
    void* p = base + off;
    off += (bytes + 255) & ~(size_t)255;
    return p;
  };
  bf16*  fbf    = (bf16*) take((size_t)EDGES * 32 * sizeof(bf16));   // 16 MB
  float* dhat   = (float*)take((size_t)EDGES * 4 * sizeof(float));   //  4 MB
  float* xf     = (float*)take((size_t)MROWS * F * sizeof(float));
  bf16*  xbf    = (bf16*) take((size_t)MROWS * F * sizeof(bf16));
  float* y      = (float*)take((size_t)MROWS * F * sizeof(float));
  bf16*  cbf    = (bf16*) take((size_t)MROWS * F * sizeof(bf16));
  bf16*  tbf    = (bf16*) take((size_t)MROWS * F * sizeof(bf16));
  float* vres   = (float*)take((size_t)MROWS * F * sizeof(float));
  bf16*  qbf    = (bf16*) take((size_t)MROWS * F * sizeof(bf16));
  bf16*  kbf    = (bf16*) take((size_t)MROWS * F * sizeof(bf16));
  bf16*  vvT    = (bf16*) take((size_t)MROWS * F * sizeof(bf16));
  float* logits = (float*)take((size_t)BB * AA * AA * sizeof(float)); // 4 MB
  bf16*  Pbf    = (bf16*) take((size_t)BB * AA * AA * sizeof(bf16));  // 2 MB
  bf16*  WfsT   = (bf16*) take((size_t)NL * F * 32 * sizeof(bf16));
  bf16*  WfpT   = (bf16*) take((size_t)NL * F * 32 * sizeof(bf16));
  bf16*  WinT   = (bf16*) take((size_t)NL * F * F * sizeof(bf16));
  bf16*  Wf2oT  = (bf16*) take((size_t)NL * F * F * sizeof(bf16));
  bf16*  WdT    = (bf16*) take((size_t)NL * F * F * sizeof(bf16));
  bf16*  WqT    = (bf16*) take((size_t)NL * F * F * sizeof(bf16));
  bf16*  WkT    = (bf16*) take((size_t)NL * F * F * sizeof(bf16));
  bf16*  WvT    = (bf16*) take((size_t)NL * F * F * sizeof(bf16));
  if (ws_size < off) return;   // scratch too small: bail deterministically

  // --- one-time prep (weights, embedding, geometry/basis) ---
  const int sqBlocks = (NL * F * F) / 256;   // 192
  prep_sq_kernel<<<sqBlocks, 256, 0, stream>>>(Win,  WinT);
  prep_sq_kernel<<<sqBlocks, 256, 0, stream>>>(Wf2o, Wf2oT);
  prep_sq_kernel<<<sqBlocks, 256, 0, stream>>>(Wd,   WdT);
  prep_sq_kernel<<<sqBlocks, 256, 0, stream>>>(Wq,   WqT);
  prep_sq_kernel<<<sqBlocks, 256, 0, stream>>>(Wk,   WkT);
  prep_sq_kernel<<<sqBlocks, 256, 0, stream>>>(Wv,   WvT);
  prep_fs_kernel<<<(NL * F * 32) / 256, 256, 0, stream>>>(Wfs, WfsT);
  prep_fs_kernel<<<(NL * F * 32) / 256, 256, 0, stream>>>(Wfp, WfpT);
  embed_kernel<<<MROWS, F, 0, stream>>>(emb, Z, xf, xbf);
  geom_kernel<<<EDGES / 256, 256, 0, stream>>>(positions, nmask, nbrs, dhat, fbf);

  // --- interaction layers ---
  for (int l = 0; l < NL; ++l) {
    const bf16* WinT_l  = WinT  + (size_t)l * F * F;
    const bf16* Wf2oT_l = Wf2oT + (size_t)l * F * F;
    const bf16* WdT_l   = WdT   + (size_t)l * F * F;
    const bf16* WqT_l   = WqT   + (size_t)l * F * F;
    const bf16* WkT_l   = WkT   + (size_t)l * F * F;
    const bf16* WvT_l   = WvT   + (size_t)l * F * F;
    const bf16* WfsT_l  = WfsT  + (size_t)l * F * 32;
    const bf16* WfpT_l  = WfpT  + (size_t)l * F * 32;

    // y = x @ Win
    gemm_feat_kernel<<<MROWS / 16, 256, 0, stream>>>(xbf, WinT_l, nullptr, nullptr, 0,
                                                     y, nullptr, nullptr);
    // fused angular CFConv -> c = cs + cp (bf16 for the next GEMM)
    edge_local_kernel<<<MROWS, 256, 0, stream>>>(fbf, dhat, nbrs, y, WfsT_l, WfpT_l, cbf);
    // t = ssp(c @ Wf2o + b)
    gemm_feat_kernel<<<MROWS / 16, 256, 0, stream>>>(cbf, Wf2oT_l, bf2o + l * F, nullptr, 1,
                                                     nullptr, tbf, nullptr);
    // v = t @ Wd + b
    gemm_feat_kernel<<<MROWS / 16, 256, 0, stream>>>(tbf, WdT_l, bd + l * F, nullptr, 0,
                                                     vres, nullptr, nullptr);
    // q,k,vv = ssp((mask*x) @ W)  (mask folded as output row scale)
    gemm_feat_kernel<<<MROWS / 16, 256, 0, stream>>>(xbf, WqT_l, nullptr, amask, 1,
                                                     nullptr, qbf, nullptr);
    gemm_feat_kernel<<<MROWS / 16, 256, 0, stream>>>(xbf, WkT_l, nullptr, amask, 1,
                                                     nullptr, kbf, nullptr);
    gemm_feat_kernel<<<MROWS / 16, 256, 0, stream>>>(xbf, WvT_l, nullptr, amask, 1,
                                                     nullptr, nullptr, vvT);
    // attention
    att_logits_kernel<<<dim3(32, BB), 256, 0, stream>>>(qbf, kbf, amask, logits);
    softmax_kernel<<<MROWS, 256, 0, stream>>>(logits, Pbf);
    float* xo = (l == NL - 1) ? out : xf;
    att_out_kernel<<<dim3(16, BB), 256, 0, stream>>>(Pbf, vvT, amask, xf, vres, xo, xbf);
  }
}
